// TimeRecurrentCell_13872744366145
// MI455X (gfx1250) — matrix operand order repaired and checked by reading, not compiled, not run on hardware
//
#include <hip/hip_runtime.h>

// LSTM: T=2048, B=32, D=512, H=512. Fused recurrent WMMA kernel, v3:
//  - 32 persistent workgroups (one 16-col j-tile of H each), 8 waves (wave32) each.
//  - wave = (gate, row-half): one 16x16 f32 accumulator tile.
//  - Critical path per step = 16 chained v_wmma_f32_16x16x32_bf16 (h-part, K=512)
//    with W_hh B-operands pinned in VGPRs (128 VGPRs, verified no spill).
//  - x-part (x_{t+1} @ W_ih^T) computed after barrier arrive, overlapping the wait.
//  - x[t+2] warmed into L2 via global_prefetch_b8 one step ahead of its use.
//  - c-state in VGPRs for all 2048 steps; h ping-pongs through a global bf16
//    buffer (L2-resident); grid-wide monotonic atomic barrier per step.

#define T_LEN 2048
#define BB    32
#define DD    512
#define HH    512
#define NWG   32            // HH / 16 column tiles

typedef __attribute__((ext_vector_type(16))) __bf16 v16bf;
typedef __attribute__((ext_vector_type(8)))  __bf16 v8bf;
typedef __attribute__((ext_vector_type(8)))  float  v8f;
typedef __attribute__((ext_vector_type(4)))  float  v4f;

// A-operand (16-bit 16x32): lane half h holds K {16h..16h+15} (contiguous, matches B).
__device__ __forceinline__ v16bf a_from_f32(const float* __restrict__ p) {
  v4f f0 = *(const v4f*)(p);
  v4f f1 = *(const v4f*)(p + 4);
  v4f f2 = *(const v4f*)(p + 8);
  v4f f3 = *(const v4f*)(p + 12);
  v16bf a;
#pragma unroll
  for (int i = 0; i < 4; ++i) {
    a[i]      = (__bf16)f0[i];
    a[4 + i]  = (__bf16)f1[i];
    a[8 + i]  = (__bf16)f2[i];
    a[12 + i] = (__bf16)f3[i];
  }
  return a;
}

__device__ __forceinline__ v16bf a_from_bf16(const __bf16* __restrict__ p) {
  v8bf lo = *(const v8bf*)(p);
  v8bf hi = *(const v8bf*)(p + 8);
  return __builtin_shufflevector(lo, hi, 0,1,2,3,4,5,6,7,8,9,10,11,12,13,14,15);
}

__device__ __forceinline__ float sigmoidf_fast(float x) {
  return 1.0f / (1.0f + __expf(-x));
}

// bias + x_t @ W_ih^T for this wave's 16x16 tile (no h dependence).
__device__ __forceinline__ v8f xpart_accum(const float* __restrict__ xp,
                                           const __bf16* __restrict__ wih_row,
                                           int kA, int kB, float bias) {
  v8f a;
#pragma unroll
  for (int e = 0; e < 8; ++e) a[e] = bias;
#pragma unroll
  for (int kt = 0; kt < 16; ++kt) {
    v16bf av = a_from_f32(xp + kt * 32 + kA);
    v16bf bv = *(const v16bf*)(wih_row + kt * 32 + kB);
    a = __builtin_amdgcn_wmma_f32_16x16x32_bf16(false, av, false, bv,
                                                (short)0, a, false, false);
  }
  return a;
}

// ---- prep kernels (run each launch: deterministic, graph-safe) ----

__global__ void prep_weights(const float* __restrict__ wihf,
                             const float* __restrict__ whhf,
                             __bf16* __restrict__ wih,
                             __bf16* __restrict__ whh, int nelem) {
  int i = blockIdx.x * blockDim.x + threadIdx.x;
  if (i < nelem) {
    wih[i] = (__bf16)wihf[i];
    whh[i] = (__bf16)whhf[i];
  }
}

__global__ void prep_misc(const float* __restrict__ b_ih,
                          const float* __restrict__ b_hh,
                          float* __restrict__ bsum,
                          __bf16* __restrict__ hb,      // [2][BB*HH]
                          unsigned* __restrict__ bar) {
  int i = blockIdx.x * blockDim.x + threadIdx.x;
  if (i < 2 * BB * HH) hb[i] = (__bf16)0.0f;
  if (i < 4 * HH)      bsum[i] = b_ih[i] + b_hh[i];
  if (i < 2)           bar[i] = 0u;
}

// ---- main persistent recurrent kernel ----

__global__ __launch_bounds__(256) void lstm_kernel(
    const float*  __restrict__ x,      // [T][B][D] f32
    const __bf16* __restrict__ wih,    // [4H][D] bf16 row-major
    const __bf16* __restrict__ whh,    // [4H][H] bf16 row-major
    const float*  __restrict__ bsum,   // [4H] = b_ih + b_hh
    __bf16*       __restrict__ hb,     // [2][B*H] ping-pong h (bf16)
    unsigned*     __restrict__ bar,    // [0]=arrive count, [1]=phase
    float*        __restrict__ out) {  // [T][B][H] ++ h_T[B][H] ++ c_T[B][H]
  const int j0   = blockIdx.x * 16;        // this WG's h-column tile
  const int tid  = threadIdx.x;
  const int lane = tid & 31;
  const int wave = tid >> 5;               // 0..7
  const int g    = wave >> 1;              // gate: 0=i 1=f 2=g 3=o
  const int r    = wave & 1;               // row half (batch 0-15 / 16-31)
  const int n    = lane & 15;              // tile column
  const int halo = lane >> 4;              // lane half
  const int kA   = halo * 16;              // A-operand K sub-offset
  const int kB   = halo * 16;              // B-operand K sub-offset

  const int   ba    = r * 16 + n;          // batch row for A-operand loads
  const int   wrow  = g * HH + j0 + n;     // weight row backing B column n
  const float bias  = bsum[g * HH + j0 + n];

  const float*  xrow    = x   + (size_t)ba * DD;
  const __bf16* wih_row = wih + (size_t)wrow * DD;
  const __bf16* whh_row = whh + (size_t)wrow * HH;

  // Pin this wave's W_hh B-operands in VGPRs for the whole sequence
  // (16 x v16bf = 128 VGPRs; constant across all 2048 steps).
  v16bf whhB[16];
#pragma unroll
  for (int kt = 0; kt < 16; ++kt)
    whhB[kt] = *(const v16bf*)(whh_row + kt * 32 + kB);

  __shared__ float gbuf[8][16][16];        // [gate*2+rowhalf][m][n]

  float cst[8];
  float hlast[8];
#pragma unroll
  for (int e = 0; e < 8; ++e) { cst[e] = 0.0f; hlast[e] = 0.0f; }

  // Warm L2 for x[0] and x[1] tiles this WG will read, then pre-compute
  // the x-part for t = 0 (h_{-1} = 0, so gates = xW + b at t=0).
  if (tid < 16) {
    const char* p0 = (const char*)x + ((size_t)blockIdx.x * 16 + tid) * 128;
    __builtin_prefetch(p0, 0, 1);
    __builtin_prefetch(p0 + (size_t)BB * DD * 4, 0, 1);
  }
  v8f accx = xpart_accum(xrow, wih_row, kA, kB, bias);

  for (int t = 0; t < T_LEN; ++t) {
    const __bf16* hread  = hb + (size_t)( t      & 1) * (BB * HH);
    __bf16*       hwrite = hb + (size_t)((t + 1) & 1) * (BB * HH);

    // ---- critical path: gates = accx + h_{t-1} @ W_hh^T  (16 chained WMMAs)
    v8f acc = accx;
    const __bf16* hp = hread + (size_t)ba * HH;
#pragma unroll
    for (int kt = 0; kt < 16; ++kt) {
      v16bf av = a_from_bf16(hp + kt * 32 + kA);
      acc = __builtin_amdgcn_wmma_f32_16x16x32_bf16(false, av, false, whhB[kt],
                                                    (short)0, acc, false, false);
    }

    // exchange gate tiles across waves (C/D layout: lane half -> M = 8*halo + e)
#pragma unroll
    for (int e = 0; e < 8; ++e) gbuf[wave][halo * 8 + e][n] = acc[e];
    __syncthreads();

    if (wave < 2) {                        // waves 0/1 own the elementwise + state
      const int r2 = wave;
#pragma unroll
      for (int e = 0; e < 8; ++e) {
        const int row = halo * 8 + e;
        const int b   = r2 * 16 + row;
        float iv = sigmoidf_fast(gbuf[0 * 2 + r2][row][n]);
        float fv = sigmoidf_fast(gbuf[1 * 2 + r2][row][n]);
        float gv = tanhf(gbuf[2 * 2 + r2][row][n]);
        float ov = sigmoidf_fast(gbuf[3 * 2 + r2][row][n]);
        float c  = fv * cst[e] + iv * gv;
        cst[e]   = c;
        float hv = ov * tanhf(c);
        hlast[e] = hv;
        out[(size_t)t * (BB * HH) + (size_t)b * HH + j0 + n] = hv;
        hwrite[(size_t)b * HH + j0 + n] = (__bf16)hv;
      }
    }
    __threadfence();                       // release h stores to agent scope
    __syncthreads();

    // arrive at the grid barrier first...
    if (tid == 0) {
      unsigned arrived = __hip_atomic_fetch_add(&bar[0], 1u, __ATOMIC_ACQ_REL,
                                                __HIP_MEMORY_SCOPE_AGENT);
      if (arrived == (unsigned)NWG * (unsigned)(t + 1) - 1u) {
        __hip_atomic_store(&bar[1], (unsigned)(t + 1), __ATOMIC_RELEASE,
                           __HIP_MEMORY_SCOPE_AGENT);
      }
    }

    // ...warm L2 for x[t+2] (512 lines of 128B split across the 32 WGs)...
    if (t + 2 < T_LEN && tid < 16) {
      const char* pf = (const char*)(x + (size_t)(t + 2) * (BB * DD)) +
                       ((size_t)blockIdx.x * 16 + tid) * 128;
      __builtin_prefetch(pf, 0, 1);
    }

    // ...then overlap the barrier wait with the x-part of step t+1
    // (uniform branch: no divergence around WMMA).
    if (t + 1 < T_LEN) {
      accx = xpart_accum(xrow + (size_t)(t + 1) * (BB * DD), wih_row,
                         kA, kB, bias);
    }

    while (__hip_atomic_load(&bar[1], __ATOMIC_RELAXED,
                             __HIP_MEMORY_SCOPE_AGENT) < (unsigned)(t + 1)) {
      __builtin_amdgcn_s_sleep(1);
    }
    __threadfence();                       // acquire: invalidate before reading new h
    __syncthreads();
  }

  // final state: h_T then c_T appended after outputs
  if (wave < 2) {
    const int r2 = wave;
#pragma unroll
    for (int e = 0; e < 8; ++e) {
      const int row = halo * 8 + e;
      const int b   = r2 * 16 + row;
      out[(size_t)T_LEN * (BB * HH) + (size_t)b * HH + j0 + n]             = hlast[e];
      out[(size_t)T_LEN * (BB * HH) + (size_t)(BB * HH)
          + (size_t)b * HH + j0 + n]                                       = cst[e];
    }
  }
}

extern "C" void kernel_launch(void* const* d_in, const int* in_sizes, int n_in,
                              void* d_out, int out_size, void* d_ws, size_t ws_size,
                              hipStream_t stream) {
  (void)in_sizes; (void)n_in; (void)out_size; (void)ws_size;
  const float* x   = (const float*)d_in[0];   // [T,B,D]
  const float* Wih = (const float*)d_in[1];   // [4H,D]
  const float* Whh = (const float*)d_in[2];   // [4H,H]
  const float* bih = (const float*)d_in[3];   // [4H]
  const float* bhh = (const float*)d_in[4];   // [4H]
  float* out = (float*)d_out;

  // workspace layout (~4.3 MB): bf16 weights, bias sum, h ping-pong, barrier
  __bf16*   wih_bf = (__bf16*)d_ws;                       // 4H*D = 1,048,576 elems
  __bf16*   whh_bf = wih_bf + (size_t)4 * HH * DD;        // 4H*H
  float*    bsum   = (float*)(whh_bf + (size_t)4 * HH * HH);
  __bf16*   hb     = (__bf16*)(bsum + 4 * HH);            // [2][B*H]
  unsigned* bar    = (unsigned*)(hb + 2 * BB * HH);

  const int nw = 4 * HH * DD;                             // 1,048,576
  prep_weights<<<(nw + 255) / 256, 256, 0, stream>>>(Wih, Whh, wih_bf, whh_bf, nw);
  prep_misc<<<(2 * BB * HH + 255) / 256, 256, 0, stream>>>(bih, bhh, bsum, hb, bar);
  lstm_kernel<<<NWG, 256, 0, stream>>>(x, wih_bf, whh_bf, bsum, hb, bar, out);
}
